// LocalContext_NO_42949673367
// MI455X (gfx1250) — matrix operand order repaired and checked by reading, not compiled, not run on hardware
//
#include <hip/hip_runtime.h>

typedef __attribute__((ext_vector_type(16))) __bf16 v16bf;
typedef __attribute__((ext_vector_type(8)))  float  v8f;

#define DEVI static __device__ __forceinline__

DEVI unsigned short f2bf(float x) {
    unsigned int u = __float_as_uint(x);
    u += 0x7FFFu + ((u >> 16) & 1u);        // round-to-nearest-even
    return (unsigned short)(u >> 16);
}
DEVI float bf2f(unsigned short h) { return __uint_as_float(((unsigned int)h) << 16); }
DEVI unsigned lds_lo32(const void* p) { return (unsigned)(unsigned long long)p; }

union FragU { uint4 q[2]; unsigned short u[16]; v16bf v; };

// ---------------------------------------------------------------- cast f32 -> bf16
__global__ void cast_bf16_kernel(const float* __restrict__ s, unsigned short* __restrict__ d, int n) {
    int i = blockIdx.x * blockDim.x + threadIdx.x;
    if (i < n) d[i] = f2bf(s[i]);
}

// ---------------------------------------------------------------- LayerNorm over C=256, x is (B,C,H,W)
__global__ void ln1_kernel(const float* __restrict__ x, const float* __restrict__ g,
                           const float* __restrict__ b, unsigned short* __restrict__ out) {
    const int C = 256, L = 1600;
    int tok = blockIdx.x;               // b*L + l
    int bb = tok / L, l = tok - bb * L;
    int c = threadIdx.x;
    float v = x[((size_t)bb * C + c) * L + l];
    __shared__ float red[256];
    red[c] = v; __syncthreads();
    for (int s = 128; s > 0; s >>= 1) { if (c < s) red[c] += red[c + s]; __syncthreads(); }
    float mu = red[0] * (1.0f / C); __syncthreads();
    float d = v - mu;
    red[c] = d * d; __syncthreads();
    for (int s = 128; s > 0; s >>= 1) { if (c < s) red[c] += red[c + s]; __syncthreads(); }
    float rs = rsqrtf(red[0] * (1.0f / C) + 1e-5f);
    out[(size_t)tok * C + c] = f2bf(d * rs * g[c] + b[c]);
}

// ---------------------------------------------------------------- LayerNorm over C2=512, y is bf16 [M,512]
__global__ void ln2_kernel(const unsigned short* __restrict__ y, const float* __restrict__ g,
                           const float* __restrict__ b, unsigned short* __restrict__ out) {
    const int C = 512;
    size_t tok = blockIdx.x;
    int t = threadIdx.x;
    float v0 = bf2f(y[tok * C + t]);
    float v1 = bf2f(y[tok * C + t + 256]);
    __shared__ float red[256];
    red[t] = v0 + v1; __syncthreads();
    for (int s = 128; s > 0; s >>= 1) { if (t < s) red[t] += red[t + s]; __syncthreads(); }
    float mu = red[0] * (1.0f / C); __syncthreads();
    float d0 = v0 - mu, d1 = v1 - mu;
    red[t] = d0 * d0 + d1 * d1; __syncthreads();
    for (int s = 128; s > 0; s >>= 1) { if (t < s) red[t] += red[t + s]; __syncthreads(); }
    float rs = rsqrtf(red[0] * (1.0f / C) + 1e-5f);
    out[tok * C + t]       = f2bf(d0 * rs * g[t] + b[t]);
    out[tok * C + t + 256] = f2bf(d1 * rs * g[t + 256] + b[t + 256]);
}

// ---------------------------------------------------------------- bf16 WMMA GEMM: C = A[M,K] x B[K,N] + bias
// Block tile 128x128, BK=32, 8 waves (4x2), wave tile 32x64 (2x4 WMMA tiles).
// A tile staged with GLOBAL_LOAD_ASYNC_TO_LDS_B128 (ASYNCcnt); B tile via register
// transpose so B-fragments are contiguous ds_load_b128. Double-buffered pipeline.
// EPI 0/1: store bf16; EPI 2: GELU(exact) then bf16; EPI 3: + residual(bf16) -> f32 NCHW out
template <int EPI>
__global__ __launch_bounds__(256)
void gemm_bf16_kernel(const unsigned short* __restrict__ A,
                      const unsigned short* __restrict__ Bw,
                      const float* __restrict__ bias,
                      const unsigned short* __restrict__ res,
                      unsigned short* __restrict__ Obf,
                      float* __restrict__ Of,
                      int M, int N, int K) {
    const int BM = 128, BN = 128, BK = 32;
    __shared__ __align__(16) unsigned short As[2][BM * BK]; // [m][k]
    __shared__ __align__(16) unsigned short Bs[2][BN * BK]; // [n][k] (transposed)
    int tid = threadIdx.x;
    int lane = tid & 31, wave = tid >> 5;
    int wm = wave & 3, wn = wave >> 2;                      // 4 waves along M, 2 along N
    int m0 = blockIdx.x * BM, n0 = blockIdx.y * BN;
    int mlo = lane & 15, grp = lane >> 4;

    v8f acc[2][4] = {};
    uint4 breg[2];

    // async stage of A tile (128x32 bf16 = 512 x 16B), 2 chunks per thread
    auto stageA = [&](int k0, int buf) {
        #pragma unroll
        for (int ii = 0; ii < 2; ii++) {
            int i = tid + ii * 256;
            int r = i >> 2, c = (i & 3) * 8;
            unsigned lds = lds_lo32(&As[buf][r * BK + c]);
            unsigned voff = (unsigned)(((size_t)(m0 + r) * K + k0 + c) * 2);
            asm volatile("global_load_async_to_lds_b128 %0, %1, %2"
                         :: "v"(lds), "v"(voff), "s"(A) : "memory");
        }
    };
    // B tile 32x128: load 2 x 16B per thread into regs
    auto loadB = [&](int k0) {
        #pragma unroll
        for (int ii = 0; ii < 2; ii++) {
            int i = tid + ii * 256;
            int kr = i >> 4, nc = (i & 15) * 8;
            breg[ii] = *(const uint4*)(&Bw[(size_t)(k0 + kr) * N + n0 + nc]);
        }
    };
    // scatter-transpose B regs into LDS [n][k]
    auto storeB = [&](int buf) {
        #pragma unroll
        for (int ii = 0; ii < 2; ii++) {
            int i = tid + ii * 256;
            int kr = i >> 4, nc = (i & 15) * 8;
            const unsigned short* ts = (const unsigned short*)&breg[ii];
            #pragma unroll
            for (int j = 0; j < 8; j++) Bs[buf][(nc + j) * BK + kr] = ts[j];
        }
    };

    int nk = K / BK;
    stageA(0, 0);
    loadB(0);
    storeB(0);
    asm volatile("s_wait_asynccnt 0x0" ::: "memory");
    __syncthreads();

    int cur = 0;
    for (int kt = 0; kt < nk; kt++) {
        if (kt + 1 < nk) {                     // issue next tile's loads before compute
            stageA((kt + 1) * BK, cur ^ 1);
            loadB((kt + 1) * BK);
        }

        FragU af[2], bfm[4];
        #pragma unroll
        for (int mt = 0; mt < 2; mt++) {
            int row = wm * 32 + mt * 16 + mlo;
            int kb = grp * 8;
            af[mt].q[0] = *(const uint4*)(&As[cur][row * BK + kb]);       // K kb..kb+7
            af[mt].q[1] = *(const uint4*)(&As[cur][row * BK + kb + 16]);  // K kb+16..kb+23
        }
        #pragma unroll
        for (int nt = 0; nt < 4; nt++) {
            int col = wn * 64 + nt * 16 + mlo;
            int kb = grp * 8;
            bfm[nt].q[0] = *(const uint4*)(&Bs[cur][col * BK + kb]);      // K kb..kb+7
            bfm[nt].q[1] = *(const uint4*)(&Bs[cur][col * BK + kb + 16]); // K kb+16..kb+23
        }
        #pragma unroll
        for (int mt = 0; mt < 2; mt++)
            #pragma unroll
            for (int nt = 0; nt < 4; nt++)
                acc[mt][nt] = __builtin_amdgcn_wmma_f32_16x16x32_bf16(
                    false, af[mt].v, false, bfm[nt].v, (short)0, acc[mt][nt], false, false);

        if (kt + 1 < nk) {
            storeB(cur ^ 1);
            asm volatile("s_wait_asynccnt 0x0" ::: "memory");
        }
        __syncthreads();
        cur ^= 1;
    }

    #pragma unroll
    for (int mt = 0; mt < 2; mt++)
        #pragma unroll
        for (int nt = 0; nt < 4; nt++) {
            int cg = n0 + wn * 64 + nt * 16 + mlo;
            float bv = bias[cg];
            #pragma unroll
            for (int vv = 0; vv < 8; vv++) {
                int rg = m0 + wm * 32 + mt * 16 + grp * 8 + vv;
                float val = acc[mt][nt][vv] + bv;
                if (EPI == 0 || EPI == 1) {
                    Obf[(size_t)rg * N + cg] = f2bf(val);
                } else if (EPI == 2) {
                    float t = 0.5f * val * (1.0f + erff(val * 0.70710678118654752f));
                    Obf[(size_t)rg * N + cg] = f2bf(t);
                } else {
                    val += bf2f(res[(size_t)rg * N + cg]);
                    int bb = rg / 1600, l = rg - bb * 1600;
                    Of[((size_t)bb * 512 + cg) * 1600 + l] = val;     // (B, 2C, H, W)
                }
            }
        }
}

// ---------------------------------------------------------------- windowed attention
DEVI int pix_l(int t, int u, int v) { return (u * 5 + t / 5) * 40 + (v * 5 + t % 5); }

__global__ void attn_kernel(const unsigned short* __restrict__ qkv,   // [B*L, 768] bf16
                            const float* __restrict__ rel_table,      // [81, 8]
                            unsigned short* __restrict__ out) {       // [B*L, 256] bf16, c = h*32+d
    const int L = 1600, W3 = 768;
    int bidx = blockIdx.x;                 // b*64 + n
    int b = bidx >> 6, n = bidx & 63;
    int u = n >> 3, v = n & 7;
    int tid = threadIdx.x, wave = tid >> 5, lane = tid & 31;
    int h = wave;                          // one wave per head
    int mlo = lane & 15, grp = lane >> 4;

    __shared__ float S[8][32][32];                           // logits (f32)
    __shared__ __align__(16) unsigned short P[8][32][32];    // probs (bf16)

    // ---- S = Q K^T (T=25 padded to 32, K=hd=32) ----
    v8f sacc[2][2] = {};
    FragU qa[2], kbm[2];
    #pragma unroll
    for (int mt = 0; mt < 2; mt++) {
        int t = mt * 16 + mlo; if (t > 24) t = 0;
        size_t rb = ((size_t)b * L + pix_l(t, u, v)) * W3;
        int kb = grp * 8;
        #pragma unroll
        for (int hh = 0; hh < 16; hh++) {
            int d = kb + hh + ((hh >= 8) ? 8 : 0);
            qa[mt].u[hh] = qkv[rb + d * 8 + h];              // q at col d*nh + h
        }
    }
    #pragma unroll
    for (int st = 0; st < 2; st++) {
        int s = st * 16 + mlo; if (s > 24) s = 0;
        size_t rb = ((size_t)b * L + pix_l(s, u, v)) * W3 + 256;
        int kb = grp * 8;
        #pragma unroll
        for (int hh = 0; hh < 16; hh++)
            kbm[st].u[hh] = qkv[rb + (kb + hh + ((hh >= 8) ? 8 : 0)) * 8 + h];     // k at col 256 + d*nh + h
    }
    #pragma unroll
    for (int mt = 0; mt < 2; mt++)
        #pragma unroll
        for (int st = 0; st < 2; st++)
            sacc[mt][st] = __builtin_amdgcn_wmma_f32_16x16x32_bf16(
                false, qa[mt].v, false, kbm[st].v, (short)0, sacc[mt][st], false, false);

    #pragma unroll
    for (int mt = 0; mt < 2; mt++)
        #pragma unroll
        for (int st = 0; st < 2; st++)
            #pragma unroll
            for (int vv = 0; vv < 8; vv++)
                S[h][mt * 16 + grp * 8 + vv][st * 16 + mlo] = sacc[mt][st][vv];
    __syncthreads();

    // ---- softmax over s (25 valid), + rel bias + checkerboard mask, scale=1/sqrt(32) ----
    {
        int t = lane;
        if (t < 25) {
            int ti = t / 5, tj = t % 5;
            int on_t = ((u * 5 + ti) + (v * 5 + tj)) & 1;
            float pv[25];
            float mx = -1e30f;
            #pragma unroll
            for (int s = 0; s < 25; s++) {
                int si = s / 5, sj = s % 5;
                int idx = (ti - si + 4) * 9 + (tj - sj + 4);
                int on_s = ((u * 5 + si) + (v * 5 + sj)) & 1;
                float m = (on_t & on_s) ? 0.0f : -100.0f;
                float val = S[h][t][s] * 0.17677669529663687f + rel_table[idx * 8 + h] + m;
                pv[s] = val; mx = fmaxf(mx, val);
            }
            float sum = 0.0f;
            #pragma unroll
            for (int s = 0; s < 25; s++) { pv[s] = __expf(pv[s] - mx); sum += pv[s]; }
            float inv = 1.0f / sum;
            #pragma unroll
            for (int s = 0; s < 25; s++) P[h][t][s] = f2bf(pv[s] * inv);
            #pragma unroll
            for (int s = 25; s < 32; s++) P[h][t][s] = 0;    // zero-pad K columns
        } else {
            #pragma unroll
            for (int s = 0; s < 32; s++) P[h][t][s] = 0;     // zero-pad M rows
        }
    }
    __syncthreads();

    // ---- O = P V (M=T pad 32, N=hd=32, K=s pad 32) ----
    v8f oacc[2][2] = {};
    FragU pa[2], vbm[2];
    #pragma unroll
    for (int mt = 0; mt < 2; mt++) {
        int t = mt * 16 + mlo;
        int kb = grp * 8;
        pa[mt].q[0] = *(const uint4*)(&P[h][t][kb]);
        pa[mt].q[1] = *(const uint4*)(&P[h][t][kb + 16]);
    }
    #pragma unroll
    for (int dt = 0; dt < 2; dt++) {
        int d = dt * 16 + mlo;
        int kb = grp * 8;
        #pragma unroll
        for (int hh = 0; hh < 16; hh++) {
            int s = kb + hh + ((hh >= 8) ? 8 : 0); if (s > 24) s = 0;              // P=0 there, keep finite
            vbm[dt].u[hh] = qkv[((size_t)b * L + pix_l(s, u, v)) * W3 + 512 + d * 8 + h];
        }
    }
    #pragma unroll
    for (int mt = 0; mt < 2; mt++)
        #pragma unroll
        for (int dt = 0; dt < 2; dt++)
            oacc[mt][dt] = __builtin_amdgcn_wmma_f32_16x16x32_bf16(
                false, pa[mt].v, false, vbm[dt].v, (short)0, oacc[mt][dt], false, false);

    // ---- store, output channel = h*32 + d (head-major) ----
    #pragma unroll
    for (int mt = 0; mt < 2; mt++)
        #pragma unroll
        for (int dt = 0; dt < 2; dt++)
            #pragma unroll
            for (int vv = 0; vv < 8; vv++) {
                int t = mt * 16 + grp * 8 + vv;
                if (t < 25) {
                    int l = pix_l(t, u, v);
                    int d = dt * 16 + mlo;
                    out[((size_t)b * L + l) * 256 + h * 32 + d] = f2bf(oacc[mt][dt][vv]);
                }
            }
}

// ----------------------------------------------------------------
extern "C" void kernel_launch(void* const* d_in, const int* in_sizes, int n_in,
                              void* d_out, int out_size, void* d_ws, size_t ws_size,
                              hipStream_t stream) {
    (void)in_sizes; (void)n_in; (void)out_size; (void)ws_size;
    const float* x        = (const float*)d_in[0];
    const float* qkv_w    = (const float*)d_in[1];
    const float* qkv_b    = (const float*)d_in[2];
    const float* rel_tab  = (const float*)d_in[3];
    const float* proj_w   = (const float*)d_in[4];
    const float* proj_b   = (const float*)d_in[5];
    const float* mlp_w1   = (const float*)d_in[6];
    const float* mlp_b1   = (const float*)d_in[7];
    const float* mlp_w2   = (const float*)d_in[8];
    const float* mlp_b2   = (const float*)d_in[9];
    const float* n1g      = (const float*)d_in[10];
    const float* n1b      = (const float*)d_in[11];
    const float* n2g      = (const float*)d_in[12];
    const float* n2b      = (const float*)d_in[13];
    float* outp = (float*)d_out;

    const int M = 64 * 1600;   // 102400 tokens

    char* ws = (char*)d_ws;
    size_t off = 0;
    auto alloc = [&](size_t bytes) -> void* {
        void* p = ws + off;
        off += (bytes + 255) & ~(size_t)255;
        return p;
    };
    unsigned short* qkvw_bf  = (unsigned short*)alloc((size_t)256 * 768 * 2);
    unsigned short* projw_bf = (unsigned short*)alloc((size_t)256 * 512 * 2);
    unsigned short* w1_bf    = (unsigned short*)alloc((size_t)512 * 1024 * 2);
    unsigned short* w2_bf    = (unsigned short*)alloc((size_t)1024 * 512 * 2);
    unsigned short* xs_bf    = (unsigned short*)alloc((size_t)M * 256 * 2);  // reused for attn out
    unsigned short* qkv_bf   = (unsigned short*)alloc((size_t)M * 768 * 2);  // reused for LN2 out
    unsigned short* y_bf     = (unsigned short*)alloc((size_t)M * 512 * 2);
    unsigned short* h1_bf    = (unsigned short*)alloc((size_t)M * 1024 * 2);
    unsigned short* attn_bf  = xs_bf;   // xs dead after QKV GEMM
    unsigned short* h_bf     = qkv_bf;  // qkv dead after attention

    // 1) weights -> bf16
    cast_bf16_kernel<<<(256 * 768 + 255) / 256, 256, 0, stream>>>(qkv_w, qkvw_bf, 256 * 768);
    cast_bf16_kernel<<<(256 * 512 + 255) / 256, 256, 0, stream>>>(proj_w, projw_bf, 256 * 512);
    cast_bf16_kernel<<<(512 * 1024 + 255) / 256, 256, 0, stream>>>(mlp_w1, w1_bf, 512 * 1024);
    cast_bf16_kernel<<<(1024 * 512 + 255) / 256, 256, 0, stream>>>(mlp_w2, w2_bf, 1024 * 512);

    // 2) LN1 -> bf16 activations [M,256]
    ln1_kernel<<<M, 256, 0, stream>>>(x, n1g, n1b, xs_bf);

    // 3) QKV GEMM [M,256]x[256,768]
    gemm_bf16_kernel<0><<<dim3(M / 128, 768 / 128), 256, 0, stream>>>(
        xs_bf, qkvw_bf, qkv_b, nullptr, qkv_bf, nullptr, M, 768, 256);

    // 4) windowed attention (8 heads, one wave each)
    attn_kernel<<<64 * 64, 256, 0, stream>>>(qkv_bf, rel_tab, attn_bf);

    // 5) proj GEMM [M,256]x[256,512] -> y (residual source)
    gemm_bf16_kernel<1><<<dim3(M / 128, 512 / 128), 256, 0, stream>>>(
        attn_bf, projw_bf, proj_b, nullptr, y_bf, nullptr, M, 512, 256);

    // 6) LN2
    ln2_kernel<<<M, 256, 0, stream>>>(y_bf, n2g, n2b, h_bf);

    // 7) MLP1 + GELU [M,512]x[512,1024]
    gemm_bf16_kernel<2><<<dim3(M / 128, 1024 / 128), 256, 0, stream>>>(
        h_bf, w1_bf, mlp_b1, nullptr, h1_bf, nullptr, M, 1024, 512);

    // 8) MLP2 + residual, write f32 NCHW output [M,1024]x[1024,512]
    gemm_bf16_kernel<3><<<dim3(M / 128, 512 / 128), 256, 0, stream>>>(
        h1_bf, w2_bf, mlp_b2, y_bf, nullptr, outp, M, 512, 1024);
}
